// BERT_EMD_128849018955
// MI455X (gfx1250) — matrix operand order repaired and checked by reading, not compiled, not run on hardware
//
#include <hip/hip_runtime.h>
#include <hip/hip_bf16.h>

// ---------------------------------------------------------------------------
// BERT-EMD Sinkhorn for MI455X (gfx1250, wave32, WMMA)
//   B=64, Lq=256, Ld=512, H=768, nit from d_in[4]
// Phases:
//   1) normalize q/d rows -> bf16 Qn/Dn in d_ws
//   2) C = 1 - Qn Dn^T via v_wmma_f32_16x16x32_bf16 (f32 accum), per-batch
//      max via int-punned atomicMax
//   3) Sinkhorn: 1 WG per batch; K = exp(-C/cmax) as f16 resident in LDS
//      (256KB < 320KB WGP LDS); 100 iterations entirely out of LDS;
//      final T/dist write.
// d_ws usage: 75,497,728 bytes (bf16 Qn 25.2MB + bf16 Dn 50.3MB + cmax 256B)
// ---------------------------------------------------------------------------

#define B_   64
#define LQ_  256
#define LD_  512
#define H_   768

typedef unsigned short u16;
typedef __bf16  v8bf  __attribute__((ext_vector_type(8)));
typedef __bf16  v16bf __attribute__((ext_vector_type(16)));
typedef float   v8f   __attribute__((ext_vector_type(8)));
typedef _Float16 h2v  __attribute__((ext_vector_type(2)));

__device__ __forceinline__ float wave_sum(float x) {
  #pragma unroll
  for (int o = 16; o; o >>= 1) x += __shfl_xor(x, o, 32);
  return x;
}
__device__ __forceinline__ float wave_max(float x) {
  #pragma unroll
  for (int o = 16; o; o >>= 1) x = fmaxf(x, __shfl_xor(x, o, 32));
  return x;
}
__device__ __forceinline__ u16 f32_to_bf16(float f) {   // RNE
  unsigned int x = __float_as_uint(f);
  return (u16)((x + 0x7fffu + ((x >> 16) & 1u)) >> 16);
}

// ---------------------------------------------------------------- init cmax
__global__ void init_cmax_kernel(float* __restrict__ cmax) {
  if (threadIdx.x < B_) cmax[threadIdx.x] = 0.0f;   // C in [0,2] -> 0 is safe floor
}

// ------------------------------------------------------------- normalization
// one wave per row; rows 0..B*LQ-1 are q, rest are d
__global__ __launch_bounds__(256) void normalize_kernel(
    const float* __restrict__ q, const float* __restrict__ d,
    u16* __restrict__ qn, u16* __restrict__ dn) {
  const int w = threadIdx.x >> 5, lane = threadIdx.x & 31;
  const long row = (long)blockIdx.x * 8 + w;          // < B*(LQ+LD)
  const float* src;  u16* dst;
  if (row < (long)B_ * LQ_) { src = q + row * H_;  dst = qn + row * H_; }
  else { long r2 = row - (long)B_ * LQ_;  src = d + r2 * H_;  dst = dn + r2 * H_; }
  float ss = 0.f;
  for (int i = lane; i < H_; i += 32) { float x = src[i]; ss += x * x; }
  ss = wave_sum(ss);
  const float s = 1.0f / fmaxf(sqrtf(ss), 1e-12f);
  for (int i = lane; i < H_; i += 32) dst[i] = f32_to_bf16(src[i] * s);
}

// ---------------------------------------------------------------- WMMA GEMM
// grid (LD/64, LQ/64, B), block 512 = 16 waves, each wave one 16x16 C tile
__global__ __launch_bounds__(512) void gemm_kernel(
    const u16* __restrict__ qn, const u16* __restrict__ dn,
    float* __restrict__ Cout, float* __restrict__ cmax) {
  const int tid  = threadIdx.x;
  const int w    = tid >> 5, lane = tid & 31;
  const int wm   = w >> 2, wn = w & 3;                 // 4x4 wave grid
  const int b    = blockIdx.z;
  const int m0   = blockIdx.y * 64 + wm * 16;
  const int n0   = blockIdx.x * 64 + wn * 16;
  const int hh   = lane >> 4;                          // lane half
  const int lr   = lane & 15;

  // A: lane L holds row m0+lr; K halves 0-7/16-23 (h=0) or 8-15/24-31 (h=1)
  const u16* Arow = qn + ((size_t)b * LQ_ + m0 + lr) * H_;
  // B: lane L holds col n0+lr; K halves 0-7/16-23 (h=0) or 8-15/24-31 (h=1), Dn is N-major/K-contig
  const u16* Brow = dn + ((size_t)b * LD_ + n0 + lr) * H_;

  v8f acc = {};
  #pragma unroll
  for (int k0 = 0; k0 < H_; k0 += 32) {
    v8bf a_lo = *(const v8bf*)(Arow + k0 + hh * 8);
    v8bf a_hi = *(const v8bf*)(Arow + k0 + 16 + hh * 8);
    v8bf b_lo = *(const v8bf*)(Brow + k0 + hh * 8);
    v8bf b_hi = *(const v8bf*)(Brow + k0 + 16 + hh * 8);
    v16bf av = __builtin_shufflevector(a_lo, a_hi, 0,1,2,3,4,5,6,7,8,9,10,11,12,13,14,15);
    v16bf bv = __builtin_shufflevector(b_lo, b_hi, 0,1,2,3,4,5,6,7,8,9,10,11,12,13,14,15);
    acc = __builtin_amdgcn_wmma_f32_16x16x32_bf16(
        /*neg_a=*/false, av, /*neg_b=*/false, bv,
        /*c_mod=*/(short)0, acc, /*reuse_a=*/false, /*reuse_b=*/false);
  }

  // C/D layout: element r -> row m0 + hh*8 + r, col n0 + lr
  float* Cp = Cout + ((size_t)b * LQ_ + m0 + hh * 8) * LD_ + n0 + lr;
  float wmax = 0.f;
  #pragma unroll
  for (int r = 0; r < 8; ++r) {
    float cv = 1.0f - acc[r];
    Cp[(size_t)r * LD_] = cv;
    wmax = fmaxf(wmax, cv);
  }
  wmax = wave_max(wmax);
  if (lane == 0) atomicMax((int*)(cmax + b), __float_as_int(wmax));  // C >= 0
}

// ------------------------------------------------------------------ Sinkhorn
// one block (512 thr, 16 waves) per batch; K f16 in LDS
__global__ __launch_bounds__(512) void sinkhorn_kernel(
    const float* __restrict__ Cg, const float* __restrict__ qmask,
    const float* __restrict__ dmask, const float* __restrict__ cmax,
    const int* __restrict__ nit_p, float* __restrict__ Tg,
    float* __restrict__ dist) {
  extern __shared__ char smem[];
  _Float16* K  = (_Float16*)smem;                        // LQ*LD f16 = 256KB
  float*    u  = (float*)(smem + LQ_ * LD_ * 2);         // 256
  float*    v  = u + LQ_;                                // 512
  float*    qd = v + LD_;                                // 256
  float*    dd = qd + LQ_;                               // 512
  float*    scr = dd + LD_;                              // 512 scratch
  const h2v* Kp = (const h2v*)K;

  const int tid = threadIdx.x, w = tid >> 5, lane = tid & 31;
  const int b = blockIdx.x;
  const size_t base = (size_t)b * LQ_ * LD_;
  const int nit = *nit_p;

  // marginals
  float qs = (tid < LQ_) ? qmask[(size_t)b * LQ_ + tid] : 0.f;
  float ds = dmask[(size_t)b * LD_ + tid];
  qs = wave_sum(qs);  ds = wave_sum(ds);
  if (lane == 0) { scr[w] = qs; scr[16 + w] = ds; }
  __syncthreads();
  if (w == 0) {
    float a = (lane < 16) ? scr[lane] : 0.f;       a = wave_sum(a);
    float c = (lane < 16) ? scr[16 + lane] : 0.f;  c = wave_sum(c);
    if (lane == 0) { scr[32] = a; scr[33] = c; }
  }
  __syncthreads();
  const float qsum = scr[32], dsum = scr[33];
  if (tid < LQ_) qd[tid] = qmask[(size_t)b * LQ_ + tid] / qsum;
  dd[tid] = dmask[(size_t)b * LD_ + tid] / dsum;

  // K = exp(-C/cmax) into LDS (f16); u = 1
  const float inv = 1.0f / cmax[b];
  for (int idx = tid; idx < LQ_ * LD_; idx += 512)
    K[idx] = (_Float16)__expf(-Cg[base + idx] * inv);
  if (tid < LQ_) u[tid] = 1.0f;
  __syncthreads();

  // Sinkhorn iterations
  const int p = tid & 255;          // column pair (2p, 2p+1)
  const int h = tid >> 8;           // q-half for v-update
  for (int it = 0; it < nit; ++it) {
    // v = d_dist / (K^T u): thread handles 2 adjacent cols over half the q's
    float a0 = 0.f, a1 = 0.f;
    const int q0 = h * (LQ_ / 2);
    for (int q = q0; q < q0 + LQ_ / 2; ++q) {
      h2v k = Kp[q * (LD_ / 2) + p];
      float uq = u[q];
      a0 += (float)k.x * uq;  a1 += (float)k.y * uq;
    }
    if (h) { scr[2 * p] = a0; scr[2 * p + 1] = a1; }
    __syncthreads();
    if (!h) {
      a0 += scr[2 * p];  a1 += scr[2 * p + 1];
      v[2 * p]     = dd[2 * p]     / a0;
      v[2 * p + 1] = dd[2 * p + 1] / a1;
    }
    __syncthreads();
    // u = q_dist / (K v): wave w handles rows [w*16, w*16+16)
    #pragma unroll 1
    for (int j = 0; j < 16; ++j) {
      const int q = w * 16 + j;
      float acc = 0.f;
      #pragma unroll
      for (int i = 0; i < 8; ++i) {
        int c2 = lane + 32 * i;                 // DWORD (pair) index in row
        h2v k = Kp[q * (LD_ / 2) + c2];
        acc += (float)k.x * v[2 * c2] + (float)k.y * v[2 * c2 + 1];
      }
      acc = wave_sum(acc);
      if (lane == 0) u[q] = qd[q] / acc;
    }
    __syncthreads();
  }

  // T = u * K * v ; distances = sum(C*T)
  float da = 0.f;
  for (int idx = tid; idx < LQ_ * LD_; idx += 512) {
    const int q = idx >> 9, dc = idx & (LD_ - 1);
    float t = u[q] * (float)K[idx] * v[dc];
    Tg[base + idx] = t;
    da += Cg[base + idx] * t;
  }
  da = wave_sum(da);
  if (lane == 0) scr[w] = da;
  __syncthreads();
  if (w == 0) {
    float t = (lane < 16) ? scr[lane] : 0.f;
    t = wave_sum(t);
    if (lane == 0) dist[b] = t;
  }
}

// -------------------------------------------------------------------- launch
extern "C" void kernel_launch(void* const* d_in, const int* in_sizes, int n_in,
                              void* d_out, int out_size, void* d_ws, size_t ws_size,
                              hipStream_t stream) {
  (void)in_sizes; (void)n_in; (void)out_size; (void)ws_size;
  const float* q  = (const float*)d_in[0];
  const float* qm = (const float*)d_in[1];
  const float* d  = (const float*)d_in[2];
  const float* dm = (const float*)d_in[3];
  const int*  nit = (const int*)d_in[4];

  float* out  = (float*)d_out;
  float* dist = out;                                   // [64]
  float* C    = out + B_;                              // [64*256*512]
  float* T    = C + (size_t)B_ * LQ_ * LD_;            // [64*256*512]

  u16*   qn   = (u16*)d_ws;                            // bf16 Qn
  u16*   dn   = qn + (size_t)B_ * LQ_ * H_;            // bf16 Dn
  float* cmax = (float*)(dn + (size_t)B_ * LD_ * H_);  // [64]

  init_cmax_kernel<<<1, 64, 0, stream>>>(cmax);
  normalize_kernel<<<(B_ * (LQ_ + LD_)) / 8, 256, 0, stream>>>(q, d, qn, dn);
  gemm_kernel<<<dim3(LD_ / 64, LQ_ / 64, B_), 512, 0, stream>>>(qn, dn, C, cmax);
  const size_t smem = (size_t)LQ_ * LD_ * 2 + 2048 * 4;   // 270336 B < 320KB
  sinkhorn_kernel<<<B_, 512, smem, stream>>>(C, qm, dm, cmax, nit, T, dist);
}
